// PrototypicalHead_32349693674155
// MI455X (gfx1250) — compile-verified, matrix operand order checked by reading, not run on hardware
//
#include <hip/hip_runtime.h>
#include <hip/hip_bf16.h>

typedef __attribute__((ext_vector_type(16))) _Float16 v16h;
typedef __attribute__((ext_vector_type(8)))  _Float16 v8h;
typedef __attribute__((ext_vector_type(8)))  float    v8f;
typedef __attribute__((ext_vector_type(4)))  float    v4f;

#define DIMS 256
#define NCLS 1024
#define EPSN 1e-8f
#define MROWS 32       // query rows per block (2 WMMA M-tiles)
#define LSTRIDE 1028   // 32x1028 f32 logits tile; 8*1028 % 64 == 32 -> conflict-free spill

union ABfrag { v16h v; v8h h[2]; };

// ---------------- Kernel Z: zero segment-sum accumulators ----------------
__global__ void k_zero(float* __restrict__ p, int n) {
  int i = blockIdx.x * blockDim.x + threadIdx.x;
  int stride = gridDim.x * blockDim.x;
  for (; i < n; i += stride) p[i] = 0.0f;
}

// ---------------- Kernel S: normalize support rows, atomic segment-sum ----------------
// One wave (32 lanes) per support row; each lane owns 8 of the 256 dims.
__global__ __launch_bounds__(256)
void k_support(const float* __restrict__ sf, const int* __restrict__ lab,
               float* __restrict__ sums, float* __restrict__ counts, int Ns) {
  int wave = (int)((blockIdx.x * blockDim.x + threadIdx.x) >> 5);
  int lane = threadIdx.x & 31;
  if (wave >= Ns) return;
  const float* row = sf + (size_t)wave * DIMS + lane * 8;
  v4f a = *(const v4f*)(row);
  v4f b = *(const v4f*)(row + 4);
  float ss = a.x*a.x + a.y*a.y + a.z*a.z + a.w*a.w
           + b.x*b.x + b.y*b.y + b.z*b.z + b.w*b.w;
  #pragma unroll
  for (int off = 16; off > 0; off >>= 1) ss += __shfl_xor(ss, off, 32);
  float scale = 1.0f / fmaxf(sqrtf(ss), EPSN);
  int c = lab[wave];
  float* dst = sums + (size_t)c * DIMS + lane * 8;
  atomicAdd(dst + 0, a.x * scale);
  atomicAdd(dst + 1, a.y * scale);
  atomicAdd(dst + 2, a.z * scale);
  atomicAdd(dst + 3, a.w * scale);
  atomicAdd(dst + 4, b.x * scale);
  atomicAdd(dst + 5, b.y * scale);
  atomicAdd(dst + 6, b.z * scale);
  atomicAdd(dst + 7, b.w * scale);
  if (lane == 0) atomicAdd(counts + c, 1.0f);
}

// ---------------- Kernel P: mean + L2-normalize -> f16 prototypes (class-major) ----------------
__global__ __launch_bounds__(256)
void k_proto(const float* __restrict__ sums, const float* __restrict__ counts,
             _Float16* __restrict__ proto16) {
  int cls  = (int)((blockIdx.x * blockDim.x + threadIdx.x) >> 5);
  int lane = threadIdx.x & 31;
  if (cls >= NCLS) return;
  const float* s = sums + (size_t)cls * DIMS + lane * 8;
  float inv = 1.0f / fmaxf(counts[cls], 1.0f);
  float v[8];
  float ss = 0.0f;
  #pragma unroll
  for (int j = 0; j < 8; ++j) { v[j] = s[j] * inv; ss += v[j] * v[j]; }
  #pragma unroll
  for (int off = 16; off > 0; off >>= 1) ss += __shfl_xor(ss, off, 32);
  float scale = 1.0f / fmaxf(sqrtf(ss), EPSN);
  _Float16* d = proto16 + (size_t)cls * DIMS + lane * 8;
  #pragma unroll
  for (int j = 0; j < 8; ++j) d[j] = (_Float16)(v[j] * scale);
}

// ---------------- Kernel G: fused WMMA GEMM + log_softmax ----------------
// Block = 256 threads (8 waves). Block owns 32 query rows x all 1024 classes.
// Wave w owns classes [w*128, w*128+128): 8 N-tiles x 2 M-tiles; K-loop 256/32 = 8.
// B fragments are software-pipelined 2 steps ahead through a 3-slot register buffer.
__global__ __launch_bounds__(256)
void k_gemm_lsm(const float* __restrict__ qf, const _Float16* __restrict__ proto16,
                float* __restrict__ out, int Nq) {
  __shared__ __align__(16) _Float16 aQ[MROWS * DIMS];  // normalized f16 queries (16 KB)
  __shared__ float slog[MROWS * LSTRIDE];              // logits tile (~131.6 KB)
  __shared__ float red[MROWS * 8];
  __shared__ float rowmax[MROWS];
  __shared__ float rowlse[MROWS];

  const int tid = threadIdx.x;
  const int q0  = blockIdx.x * MROWS;
  const int r   = tid >> 3;       // query row 0..31
  const int sub = tid & 7;        // 8 threads per row

  // ---- load 32 f32 per thread, row-norm via LDS, emit f16 into aQ ----
  v4f q4[8];
  const float* qrow = qf + (size_t)(q0 + r) * DIMS + sub * 32;
  float ss = 0.0f;
  #pragma unroll
  for (int j = 0; j < 8; ++j) {
    q4[j] = *(const v4f*)(qrow + j * 4);
    ss += q4[j].x*q4[j].x + q4[j].y*q4[j].y + q4[j].z*q4[j].z + q4[j].w*q4[j].w;
  }
  red[r * 8 + sub] = ss;
  __syncthreads();
  if (sub == 0) {
    float t = 0.0f;
    #pragma unroll
    for (int j = 0; j < 8; ++j) t += red[r * 8 + j];
    rowmax[r] = 1.0f / fmaxf(sqrtf(t), EPSN);   // temporarily holds the scale
  }
  __syncthreads();
  {
    float scale = rowmax[r];
    _Float16* arow = aQ + r * DIMS + sub * 32;
    #pragma unroll
    for (int j = 0; j < 8; ++j) {
      arow[j * 4 + 0] = (_Float16)(q4[j].x * scale);
      arow[j * 4 + 1] = (_Float16)(q4[j].y * scale);
      arow[j * 4 + 2] = (_Float16)(q4[j].z * scale);
      arow[j * 4 + 3] = (_Float16)(q4[j].w * scale);
    }
  }
  __syncthreads();

  // ---- WMMA GEMM: 32 queries x 128 classes per wave, pipelined B prefetch ----
  const int lane = tid & 31;
  const int wv   = tid >> 5;            // wave 0..7
  const int m    = lane & 15;           // A row / B column within tile
  const int khi  = (lane >> 4) * 8;     // half-wave K offset per ISA 16-bit layout

  const _Float16* pbase = proto16 + (size_t)(wv * 128 + m) * DIMS + khi;

  v8f acc[2][8];
  #pragma unroll
  for (int mt = 0; mt < 2; ++mt)
    #pragma unroll
    for (int t = 0; t < 8; ++t)
      acc[mt][t] = (v8f){0.f,0.f,0.f,0.f,0.f,0.f,0.f,0.f};

  ABfrag a0, a1;
  ABfrag bbuf[3];

  // loadB(i): i = kstep*8 + tile; B row = class (wv*128 + t*16 + m), K slice kb..kb+32
  #define LOADB(i, slot)                                                         \
    do {                                                                         \
      const int kb_ = ((i) >> 3) * 32;                                           \
      const int t_  = (i) & 7;                                                   \
      const _Float16* prow_ = pbase + (size_t)(t_ * 16) * DIMS + kb_;            \
      bbuf[slot].h[0] = *(const v8h*)(prow_);                                    \
      bbuf[slot].h[1] = *(const v8h*)(prow_ + 16);                               \
    } while (0)

  LOADB(0, 0);
  LOADB(1, 1);
  #pragma unroll
  for (int i = 0; i < 64; ++i) {
    const int kb = (i >> 3) * 32;
    const int t  = i & 7;
    if (t == 0) {
      a0.h[0] = *(const v8h*)(aQ + (m)      * DIMS + kb + khi);
      a0.h[1] = *(const v8h*)(aQ + (m)      * DIMS + kb + 16 + khi);
      a1.h[0] = *(const v8h*)(aQ + (m + 16) * DIMS + kb + khi);
      a1.h[1] = *(const v8h*)(aQ + (m + 16) * DIMS + kb + 16 + khi);
    }
    if (i + 2 < 64) LOADB(i + 2, (i + 2) % 3);
    ABfrag& b = bbuf[i % 3];
    acc[0][t] = __builtin_amdgcn_wmma_f32_16x16x32_f16(
        false, a0.v, false, b.v, (short)0, acc[0][t], false, false);
    acc[1][t] = __builtin_amdgcn_wmma_f32_16x16x32_f16(
        false, a1.v, false, b.v, (short)0, acc[1][t], false, false);
  }
  #undef LOADB

  // ---- spill accumulators to LDS logits tile (C-layout: VGPR rr -> M = rr + hi*8) ----
  {
    const int hi = lane >> 4;
    #pragma unroll
    for (int mt = 0; mt < 2; ++mt)
      #pragma unroll
      for (int t = 0; t < 8; ++t) {
        const int ncol = wv * 128 + t * 16 + m;
        #pragma unroll
        for (int rr = 0; rr < 8; ++rr)
          slog[(mt * 16 + rr + hi * 8) * LSTRIDE + ncol] = acc[mt][t][rr];
      }
  }
  __syncthreads();

  // ---- log_softmax over 1024 classes per row (8 threads/row, 128 cols each) ----
  float mx = -3.402823466e+38f;
  for (int j = 0; j < 128; ++j) mx = fmaxf(mx, slog[r * LSTRIDE + sub + j * 8]);
  red[r * 8 + sub] = mx;
  __syncthreads();
  if (sub == 0) {
    float t = red[r * 8];
    #pragma unroll
    for (int j = 1; j < 8; ++j) t = fmaxf(t, red[r * 8 + j]);
    rowmax[r] = t;
  }
  __syncthreads();
  const float rm = rowmax[r];
  float se = 0.0f;
  for (int j = 0; j < 128; ++j) se += __expf(slog[r * LSTRIDE + sub + j * 8] - rm);
  red[r * 8 + sub] = se;
  __syncthreads();
  if (sub == 0) {
    float t = 0.0f;
    #pragma unroll
    for (int j = 0; j < 8; ++j) t += red[r * 8 + j];
    rowlse[r] = __logf(t);
  }
  __syncthreads();

  // ---- single coalesced output write: 32 float4 per thread ----
  const float shift = rm + rowlse[r];
  float* orow = out + (size_t)(q0 + r) * NCLS + sub * 128;
  const float* lrow = slog + r * LSTRIDE + sub * 128;
  #pragma unroll
  for (int j = 0; j < 32; ++j) {
    v4f v;
    v.x = lrow[j * 4 + 0] - shift;
    v.y = lrow[j * 4 + 1] - shift;
    v.z = lrow[j * 4 + 2] - shift;
    v.w = lrow[j * 4 + 3] - shift;
    *(v4f*)(orow + j * 4) = v;
  }
}

// ---------------- Kernel T: unique_classes = arange(1024) appended to output ----------------
__global__ void k_tail(float* __restrict__ out, unsigned long long base, int n) {
  int i = blockIdx.x * blockDim.x + threadIdx.x;
  if (i < n) out[base + i] = (float)i;
}

extern "C" void kernel_launch(void* const* d_in, const int* in_sizes, int n_in,
                              void* d_out, int out_size, void* d_ws, size_t ws_size,
                              hipStream_t stream) {
  const float* sf  = (const float*)d_in[0];
  const int*   lab = (const int*)d_in[1];
  const float* qf  = (const float*)d_in[2];
  float* out = (float*)d_out;

  const int Ns = in_sizes[0] / DIMS;
  const int Nq = in_sizes[2] / DIMS;

  // workspace layout: [sums: C*D f32][counts: C f32][proto16: C*D f16]  (~1.6 MB)
  float*     sums    = (float*)d_ws;
  float*     counts  = sums + (size_t)NCLS * DIMS;
  _Float16*  proto16 = (_Float16*)(counts + NCLS);

  // Z: zero accumulators (every call; no persistent state)
  k_zero<<<256, 256, 0, stream>>>(sums, NCLS * DIMS + NCLS);

  // S: normalized segment-sum (one wave per support row)
  k_support<<<(Ns * 32 + 255) / 256, 256, 0, stream>>>(sf, lab, sums, counts, Ns);

  // P: prototypes -> f16 (one wave per class)
  k_proto<<<(NCLS * 32 + 255) / 256, 256, 0, stream>>>(sums, counts, proto16);

  // G: fused WMMA GEMM + log_softmax (one block per 32 query rows)
  k_gemm_lsm<<<Nq / MROWS, 256, 0, stream>>>(qf, proto16, out, Nq);

  // T: arange tail for unique_classes
  long long tail = (long long)out_size - (long long)Nq * NCLS;
  if (tail > 0)
    k_tail<<<((int)tail + 255) / 256, 256, 0, stream>>>(out, (unsigned long long)Nq * NCLS, (int)tail);
}
